// VectorQuantizer_34892314313112
// MI455X (gfx1250) — compile-verified, matrix operand order checked by reading, not run on hardware
//
#include <hip/hip_runtime.h>

typedef _Float16 half8  __attribute__((ext_vector_type(8)));
typedef _Float16 half16 __attribute__((ext_vector_type(16)));
typedef float    float8 __attribute__((ext_vector_type(8)));

#define NUM_EMB 1024
#define DIM     256
#define HW      4096      // 64*64
#define NPIX    131072    // 32*4096
#define NUMEL   33554432  // 32*256*4096

// ---------------------------------------------------------------------------
// Kernel 1: codebook prep — fp32 -> f16 copy + per-entry squared norms
// grid = 1024 (one block per entry), block = 256 (one thread per dim)
// ---------------------------------------------------------------------------
__global__ __launch_bounds__(256) void vq_prep(const float* __restrict__ emb,
                                               _Float16* __restrict__ embh,
                                               float* __restrict__ norms) {
    const int e = blockIdx.x;
    const int t = threadIdx.x;
    const float v = emb[e * DIM + t];
    embh[e * DIM + t] = (_Float16)v;
    float s = v * v;
#pragma unroll
    for (int m = 16; m >= 1; m >>= 1) s += __shfl_xor(s, m, 32);
    __shared__ float wsum[8];
    const int lane = t & 31, w = t >> 5;
    if (lane == 0) wsum[w] = s;
    __syncthreads();
    if (t == 0) {
        float tot = 0.f;
#pragma unroll
        for (int i = 0; i < 8; i++) tot += wsum[i];
        norms[e] = tot;
    }
}

// ---------------------------------------------------------------------------
// Kernel 2: WMMA distance matmul + row argmin over 1024 codebook entries.
// block = 256 threads = 8 waves; each wave owns TWO 16-row A tiles (32 pixel
// rows), reusing each B register tile for two WMMAs (halves B cache traffic,
// loads:wmma ratio 1:1).  grid = NPIX/256 = 512 blocks.
// ---------------------------------------------------------------------------
__global__ __launch_bounds__(256) void vq_argmin(const float* __restrict__ z,
                                                 const _Float16* __restrict__ embh,
                                                 const float* __restrict__ norms,
                                                 float* __restrict__ idxF) {
    const int lane = threadIdx.x & 31;
    const int wave = threadIdx.x >> 5;
    const int rowBase = blockIdx.x * 256 + wave * 32;
    const int r16 = lane & 15;               // row (A) / entry-col (B) within tile
    const int hi  = (lane >> 4) & 1;         // lane half selects K sub-range

    // Preload A: two 16x256 tiles of z in WMMA 16-bit A-layout.
    half16 a[2][8];
#pragma unroll
    for (int m = 0; m < 2; m++) {
        const int n_row = rowBase + m * 16 + r16;
        const int bimg  = n_row >> 12;
        const int hw    = n_row & (HW - 1);
        const size_t zbase = (size_t)bimg * ((size_t)DIM * HW) + (size_t)hw;
#pragma unroll
        for (int c = 0; c < 8; c++) {
            const int d0 = c * 32 + hi * 8;
#pragma unroll
            for (int i = 0; i < 8; i++) {
                a[m][c][i]     = (_Float16)z[zbase + (size_t)(d0 + i) * HW];
                a[m][c][i + 8] = (_Float16)z[zbase + (size_t)(d0 + 16 + i) * HW];
            }
        }
    }

    float best[2][8];
    int   bidx[2][8];
#pragma unroll
    for (int m = 0; m < 2; m++)
#pragma unroll
        for (int r = 0; r < 8; r++) { best[m][r] = 3.4e38f; bidx[m][r] = 0; }

    for (int t = 0; t < NUM_EMB / 16; t++) {
        const int e = t * 16 + r16;                 // codebook entry this lane covers
        const float nrm = norms[e];
        const _Float16* ep = embh + (size_t)e * DIM + hi * 8;
        // Warm WGP$ with next tile's entry row while this tile's WMMAs run.
        if (t < NUM_EMB / 16 - 1) __builtin_prefetch(ep + 16 * DIM, 0, 0);
        float8 acc0 = {};
        float8 acc1 = {};
#pragma unroll
        for (int c = 0; c < 8; c++) {
            half8 blo = *(const half8*)(ep + c * 32);        // K = d0 .. d0+7
            half8 bhi = *(const half8*)(ep + c * 32 + 16);   // K = d0+16 .. d0+23
            half16 b = __builtin_shufflevector(blo, bhi,
                        0,1,2,3,4,5,6,7,8,9,10,11,12,13,14,15);
            acc0 = __builtin_amdgcn_wmma_f32_16x16x32_f16(
                       false, a[0][c], false, b, (short)0, acc0, false, false);
            acc1 = __builtin_amdgcn_wmma_f32_16x16x32_f16(
                       false, a[1][c], false, b, (short)0, acc1, false, false);
        }
        // dist = ||e||^2 - 2 z.e   (||z||^2 dropped: constant per row)
#pragma unroll
        for (int r = 0; r < 8; r++) {
            const float s0 = nrm - 2.0f * acc0[r];
            if (s0 < best[0][r]) { best[0][r] = s0; bidx[0][r] = e; }
            const float s1 = nrm - 2.0f * acc1[r];
            if (s1 < best[1][r]) { best[1][r] = s1; bidx[1][r] = e; }
        }
    }

    // Cross-lane min+argmin within each 16-lane half (C/D layout: lanes 0-15
    // hold M=r, lanes 16-31 hold M=r+8; masks 1..8 stay inside the half).
#pragma unroll
    for (int m = 0; m < 2; m++) {
#pragma unroll
        for (int r = 0; r < 8; r++) {
#pragma unroll
            for (int msk = 1; msk < 16; msk <<= 1) {
                const float ov = __shfl_xor(best[m][r], msk, 32);
                const int   oi = __shfl_xor(bidx[m][r], msk, 32);
                if (ov < best[m][r] || (ov == best[m][r] && oi < bidx[m][r])) {
                    best[m][r] = ov; bidx[m][r] = oi;
                }
            }
        }
    }
    if (r16 == 0) {
#pragma unroll
        for (int m = 0; m < 2; m++)
#pragma unroll
            for (int r = 0; r < 8; r++) {
                const int pix = rowBase + m * 16 + hi * 8 + r;
                idxF[pix] = (float)bidx[m][r];
            }
    }
}

// ---------------------------------------------------------------------------
// Kernel 3: gather + straight-through output + per-block partial MSE sums.
// grid = NUMEL/256 blocks; consecutive threads -> consecutive hw (coalesced
// z read / out write); codebook gathers hit L2/WGP$.
// ---------------------------------------------------------------------------
__global__ __launch_bounds__(256) void vq_gather(const float* __restrict__ z,
                                                 const float* __restrict__ emb,
                                                 const float* __restrict__ idxF,
                                                 float* __restrict__ outQ,
                                                 float* __restrict__ partials) {
    const unsigned o  = blockIdx.x * 256u + threadIdx.x;
    const unsigned b  = o >> 20;           // / (DIM*HW) = 2^20
    const unsigned d  = (o >> 12) & 255u;
    const unsigned hw = o & 4095u;
    const unsigned n  = (b << 12) | hw;
    const int idx = (int)idxF[n];
    const float q  = emb[(size_t)idx * DIM + d];
    const float zv = z[o];
    outQ[o] = zv + (q - zv);               // straight-through, same fp ops as ref
    const float diff = q - zv;
    float s = diff * diff;
#pragma unroll
    for (int m = 16; m >= 1; m >>= 1) s += __shfl_xor(s, m, 32);
    __shared__ float wsum[8];
    const int lane = threadIdx.x & 31, w = threadIdx.x >> 5;
    if (lane == 0) wsum[w] = s;
    __syncthreads();
    if (threadIdx.x == 0) {
        float tot = 0.f;
#pragma unroll
        for (int i = 0; i < 8; i++) tot += wsum[i];
        partials[blockIdx.x] = tot;
    }
}

// ---------------------------------------------------------------------------
// Kernel 4: deterministic reduction of NPIX partial sums -> scalar loss.
// loss = q_latent + 0.25 * e_latent = 1.25 * sum / NUMEL (values identical).
// ---------------------------------------------------------------------------
__global__ __launch_bounds__(1024) void vq_loss(const float* __restrict__ partials,
                                                float* __restrict__ lossOut) {
    __shared__ float sm[1024];
    float s = 0.f;
    for (int j = threadIdx.x; j < NPIX; j += 1024) s += partials[j];
    sm[threadIdx.x] = s;
    __syncthreads();
    for (int step = 512; step > 0; step >>= 1) {
        if (threadIdx.x < (unsigned)step) sm[threadIdx.x] += sm[threadIdx.x + step];
        __syncthreads();
    }
    if (threadIdx.x == 0) lossOut[0] = 1.25f * sm[0] / (float)NUMEL;
}

// ---------------------------------------------------------------------------
extern "C" void kernel_launch(void* const* d_in, const int* in_sizes, int n_in,
                              void* d_out, int out_size, void* d_ws, size_t ws_size,
                              hipStream_t stream) {
    const float* z   = (const float*)d_in[0];   // [32,256,64,64] fp32
    const float* emb = (const float*)d_in[1];   // [1024,256] fp32

    float* out     = (float*)d_out;
    float* outQ    = out;                 // NUMEL quantized_st values
    float* lossOut = out + NUMEL;         // 1 scalar loss
    float* idxF    = out + NUMEL + 1;     // NPIX indices (as float)

    char* ws = (char*)d_ws;
    _Float16* embh   = (_Float16*)ws;                                   // 512 KB
    float*    norms  = (float*)(ws + (size_t)NUM_EMB * DIM * 2);        //   4 KB
    float*    parts  = (float*)(ws + (size_t)NUM_EMB * DIM * 2
                                   + (size_t)NUM_EMB * 4);              // 512 KB

    vq_prep  <<<NUM_EMB,      DIM, 0, stream>>>(emb, embh, norms);
    vq_argmin<<<NPIX / 256,   256, 0, stream>>>(z, embh, norms, idxF);
    vq_gather<<<NUMEL / 256,  256, 0, stream>>>(z, emb, idxF, outQ, parts);
    vq_loss  <<<1,           1024, 0, stream>>>(parts, lossOut);
}
